// CustomSwinTransformer_11802570130322
// MI455X (gfx1250) — matrix operand order repaired and checked by reading, not compiled, not run on hardware
//
#include <hip/hip_runtime.h>
#include <hip/hip_bf16.h>
#include <math.h>

// ---------------------------------------------------------------------------
// Types
// ---------------------------------------------------------------------------
typedef _Float16 half_t;
typedef __attribute__((ext_vector_type(16))) _Float16 v16h;
typedef __attribute__((ext_vector_type(8)))  _Float16 v8h;
typedef __attribute__((ext_vector_type(8)))  float    v8f;

#define TILE_M 128
#define TILE_N 64
#define TILE_K 32

enum { EPI_F32 = 0, EPI_F16 = 1, EPI_F16_GELU = 2, EPI_F32_ACC = 3, EPI_F32_ADD_WIN = 4 };

// ---------------------------------------------------------------------------
// Block reduction helper (blockDim.x == 256)
// ---------------------------------------------------------------------------
__device__ inline float block_reduce_sum256(float v, float* s) {
    const int tid = threadIdx.x;
    s[tid] = v;
    __syncthreads();
    for (int off = 128; off > 0; off >>= 1) {
        if (tid < off) s[tid] += s[tid + off];
        __syncthreads();
    }
    float r = s[0];
    __syncthreads();
    return r;
}

// ---------------------------------------------------------------------------
// WMMA GEMM:  Out[M,N] = A[M,K](f16) * W[K,N](f32->f16) + bias, epilogue MODE.
// K must be a multiple of 32 (true for every GEMM in this network).
// 256 threads = 8 waves; block tile 128x64; wave tile 16x64 (4 WMMA accums).
// LDS tiles are stored directly in CDNA5 WMMA fragment layout so every lane
// fetches its whole v16h operand as one 32B (2x ds_load_b128) read.
//   A frag (16x32 f16): lane = (row%16) + 16*(kgrp&1), elem span 8*(kgrp>>1)
//   B frag (32x16 f16): lane = (col%16) + 16*(kgrp&1), elem span 8*(kgrp>>1)
// Edge handling: source rows/cols are CLAMPED (not zeroed) — out-of-range
// lanes only ever feed output elements that the epilogue never stores, so no
// per-iteration exec-mask branching is needed in the staging loop.
// EPI_F32_ADD_WIN fuses reverse-window-partition + reverse-shift + residual
// add into the epilogue (each window token maps to a unique pixel: race-free).
// ---------------------------------------------------------------------------
template <int MODE>
__global__ __launch_bounds__(256)
void k_gemm_f16(const half_t* __restrict__ A, const float* __restrict__ W,
                const float* __restrict__ bias, void* __restrict__ Out,
                int M, int N, int K, int Hd, int Wd, int shift)
{
    __shared__ __align__(32) half_t AsF[8][32][16];   // [row strip][lane][elem] 8KB
    __shared__ __align__(32) half_t BsF[4][32][16];   // [col tile ][lane][elem] 4KB

    const int tid  = threadIdx.x;
    const int lane = tid & 31;
    const int wave = tid >> 5;            // 0..7 -> 16-row strip
    const int bm   = blockIdx.x * TILE_M;
    const int bn   = blockIdx.y * TILE_N;

    v8f acc[4] = {v8f{}, v8f{}, v8f{}, v8f{}};

    // ---- A staging map (two 8-half row spans per thread), loop-invariant ----
    const half_t* aptr[2];
    half_t*       adst[2];
#pragma unroll
    for (int cc = 0; cc < 2; ++cc) {
        int linear = tid * 8 + cc * 2048;
        int r = linear >> 5;              // 0..127
        int c = linear & 31;              // 0,8,16,24
        int s = r >> 4, lr = r & 15, g = c >> 3;
        adst[cc] = &AsF[s][lr + ((g & 1) << 4)][(g >> 1) << 3];
        int grow = bm + r; if (grow > M - 1) grow = M - 1;   // clamp (safe)
        aptr[cc] = A + (size_t)grow * K + c;
    }
    // ---- B staging map (one 8-deep K column segment per thread) ----
    const int b_n    = tid & 63;              // local col 0..63
    const int b_kb   = (tid >> 6) * 8;        // k base: 0,8,16,24
    int gcol = bn + b_n; if (gcol > N - 1) gcol = N - 1;     // clamp (safe)
    half_t* bdst = &BsF[b_n >> 4][(b_n & 15) + (((b_kb >> 3) & 1) << 4)][(b_kb >> 4) << 3];
    const float* bptr = W + (size_t)b_kb * N + gcol;
    const size_t bstep = (size_t)TILE_K * N;

    for (int k0 = 0; k0 < K; k0 += TILE_K) {
        // stage A
        *(v8h*)adst[0] = *(const v8h*)aptr[0];
        *(v8h*)adst[1] = *(const v8h*)aptr[1];
        // stage B (f32 -> f16 convert; coalesced across the 64 columns)
        {
            v8h tmp;
#pragma unroll
            for (int e = 0; e < 8; ++e) tmp[e] = (half_t)bptr[(size_t)e * N];
            *(v8h*)bdst = tmp;
        }
        // prefetch next K tile while WMMAs run (speculative, fault-free)
        __builtin_prefetch(aptr[0] + TILE_K, 0, 3);
        __builtin_prefetch(bptr + bstep, 0, 3);
        aptr[0] += TILE_K; aptr[1] += TILE_K; bptr += bstep;
        __syncthreads();

        v16h afrag = *(const v16h*)&AsF[wave][lane][0];
#pragma unroll
        for (int t = 0; t < 4; ++t) {
            v16h bfrag = *(const v16h*)&BsF[t][lane][0];
            acc[t] = __builtin_amdgcn_wmma_f32_16x16x32_f16(
                false, afrag, false, bfrag, (short)0, acc[t], false, false);
        }
        __syncthreads();
    }

    // ---- epilogue; C/D layout: row = r + 8*(lane>>4), col = lane&15 ----
    const int m  = lane & 15;
    const int hh = lane >> 4;
#pragma unroll
    for (int t = 0; t < 4; ++t) {
        int col = bn + t * 16 + m;
        if (col >= N) continue;
        float bv = bias ? bias[col] : 0.f;
#pragma unroll
        for (int r = 0; r < 8; ++r) {
            int row = bm + wave * 16 + r + 8 * hh;
            if (row >= M) continue;
            float val = acc[t][r] + bv;
            if (MODE == EPI_F16) {
                ((half_t*)Out)[(size_t)row * N + col] = (half_t)val;
            } else if (MODE == EPI_F16_GELU) {
                float g = 0.5f * val * (1.f + erff(val * 0.70710678118654752f));
                ((half_t*)Out)[(size_t)row * N + col] = (half_t)g;
            } else if (MODE == EPI_F32_ACC) {
                ((float*)Out)[(size_t)row * N + col] += val;
            } else if (MODE == EPI_F32_ADD_WIN) {
                // row is a window token: un-window + un-shift, add into X[img,y,x,:]
                int pos = row % 49, win = row / 49;
                int nwn = Wd / 7, nhw = (Hd / 7) * nwn;
                int img = win / nhw, wl = win % nhw;
                int wy = wl / nwn, wx = wl % nwn;
                int sy = (wy * 7 + pos / 7 + shift) % Hd;
                int sx = (wx * 7 + pos % 7 + shift) % Wd;
                ((float*)Out)[(size_t)((img * Hd + sy) * Wd + sx) * N + col] += val;
            } else {
                ((float*)Out)[(size_t)row * N + col] = val;
            }
        }
    }
}

// ---------------------------------------------------------------------------
// Patch embed: 4x4 s4 conv (K=48) + bias + LayerNorm, fused. 96 thr/token.
// ---------------------------------------------------------------------------
__global__ __launch_bounds__(96)
void k_patch_embed(const float* __restrict__ imgs, const float* __restrict__ pw,
                   const float* __restrict__ pb, const float* __restrict__ g,
                   const float* __restrict__ b, float* __restrict__ X)
{
    __shared__ float sred[96];
    const float mean3[3] = {0.481f, 0.457f, 0.408f};
    const float rstd3[3] = {1.f / 0.268f, 1.f / 0.261f, 1.f / 0.275f};

    const int token = blockIdx.x;            // img*3136 + oy*56 + ox
    const int img = token / 3136;
    const int rem = token % 3136;
    const int oy = rem / 56, ox = rem % 56;
    const int c = threadIdx.x;

    float acc = pb[c];
#pragma unroll
    for (int ky = 0; ky < 4; ++ky) {
#pragma unroll
        for (int kx = 0; kx < 4; ++kx) {
            const float* pix =
                imgs + (size_t)((img * 224 + oy * 4 + ky) * 224 + ox * 4 + kx) * 3;
            const float* wv = pw + (size_t)((ky * 4 + kx) * 3) * 96 + c;
#pragma unroll
            for (int ci = 0; ci < 3; ++ci) {
                float v = (pix[ci] * (1.f / 255.f) - mean3[ci]) * rstd3[ci];
                acc += v * wv[ci * 96];
            }
        }
    }
    // LayerNorm over 96 channels
    sred[c] = acc; __syncthreads();
    if (c < 32) sred[c] += sred[c + 32] + sred[c + 64];
    __syncthreads();
    for (int off = 16; off > 0; off >>= 1) { if (c < off) sred[c] += sred[c + off]; __syncthreads(); }
    float mean = sred[0] / 96.f; __syncthreads();
    float d = acc - mean;
    sred[c] = d * d; __syncthreads();
    if (c < 32) sred[c] += sred[c + 32] + sred[c + 64];
    __syncthreads();
    for (int off = 16; off > 0; off >>= 1) { if (c < off) sred[c] += sred[c + off]; __syncthreads(); }
    float rstd = rsqrtf(sred[0] / 96.f + 1e-5f);
    X[(size_t)token * 96 + c] = d * rstd * g[c] + b[c];
}

// ---------------------------------------------------------------------------
// LN1 fused with cyclic shift + window partition -> f16 rows in window order.
// ---------------------------------------------------------------------------
__global__ __launch_bounds__(256)
void k_ln_window(const float* __restrict__ X, const float* __restrict__ g,
                 const float* __restrict__ b, half_t* __restrict__ Out,
                 int Hd, int Wd, int C, int shift)
{
    __shared__ float sred[256];
    __shared__ float vals[768];

    const int token = blockIdx.x;            // win*49 + pos
    const int pos = token % 49;
    const int win = token / 49;
    const int nw = Wd / 7;
    const int nhw = (Hd / 7) * nw;
    const int img = win / nhw;
    const int wloc = win % nhw;
    const int wy = wloc / nw, wx = wloc % nw;
    const int py = pos / 7, px = pos % 7;
    const int sy = (wy * 7 + py + shift) % Hd;
    const int sx = (wx * 7 + px + shift) % Wd;
    const float* row = X + (size_t)((img * Hd + sy) * Wd + sx) * C;

    const int tid = threadIdx.x;
    float sum = 0.f;
    for (int c = tid; c < C; c += 256) { float v = row[c]; vals[c] = v; sum += v; }
    __syncthreads();
    float mean = block_reduce_sum256(sum, sred) / (float)C;
    float var = 0.f;
    for (int c = tid; c < C; c += 256) { float d = vals[c] - mean; var += d * d; }
    var = block_reduce_sum256(var, sred) / (float)C;
    float rstd = rsqrtf(var + 1e-5f);
    half_t* orow = Out + (size_t)token * C;
    for (int c = tid; c < C; c += 256)
        orow[c] = (half_t)((vals[c] - mean) * rstd * g[c] + b[c]);
}

// ---------------------------------------------------------------------------
// Plain LN over natural token order -> f16 (LN2 / final norm).
// ---------------------------------------------------------------------------
__global__ __launch_bounds__(256)
void k_ln_plain(const float* __restrict__ X, const float* __restrict__ g,
                const float* __restrict__ b, half_t* __restrict__ Out, int C)
{
    __shared__ float sred[256];
    __shared__ float vals[768];
    const int token = blockIdx.x;
    const float* row = X + (size_t)token * C;
    const int tid = threadIdx.x;
    float sum = 0.f;
    for (int c = tid; c < C; c += 256) { float v = row[c]; vals[c] = v; sum += v; }
    __syncthreads();
    float mean = block_reduce_sum256(sum, sred) / (float)C;
    float var = 0.f;
    for (int c = tid; c < C; c += 256) { float d = vals[c] - mean; var += d * d; }
    var = block_reduce_sum256(var, sred) / (float)C;
    float rstd = rsqrtf(var + 1e-5f);
    half_t* orow = Out + (size_t)token * C;
    for (int c = tid; c < C; c += 256)
        orow[c] = (half_t)((vals[c] - mean) * rstd * g[c] + b[c]);
}

// ---------------------------------------------------------------------------
// Patch-merge gather (2x2 -> 4C) fused with LN -> f16 rows for reduction GEMM.
// ---------------------------------------------------------------------------
__global__ __launch_bounds__(256)
void k_merge_ln(const float* __restrict__ X, const float* __restrict__ g,
                const float* __restrict__ b, half_t* __restrict__ Out,
                int Hd, int Wd, int C)
{
    __shared__ float sred[256];
    __shared__ float vals[3072];
    const int token = blockIdx.x;            // img*(H/2*W/2) + y2*(W/2)+x2
    const int H2 = Hd / 2, W2 = Wd / 2;
    const int img = token / (H2 * W2);
    const int rem = token % (H2 * W2);
    const int y2 = rem / W2, x2 = rem % W2;
    const int C4 = 4 * C;
    const int tid = threadIdx.x;

    float sum = 0.f;
    for (int cc = tid; cc < C4; cc += 256) {
        int q = cc / C, ci = cc % C;
        int dy = q & 1, dx = q >> 1;         // [x0,x1,x2,x3] quadrant order
        float v = X[(size_t)((img * Hd + 2 * y2 + dy) * Wd + 2 * x2 + dx) * C + ci];
        vals[cc] = v; sum += v;
    }
    __syncthreads();
    float mean = block_reduce_sum256(sum, sred) / (float)C4;
    float var = 0.f;
    for (int cc = tid; cc < C4; cc += 256) { float d = vals[cc] - mean; var += d * d; }
    var = block_reduce_sum256(var, sred) / (float)C4;
    float rstd = rsqrtf(var + 1e-5f);
    half_t* orow = Out + (size_t)token * C4;
    for (int cc = tid; cc < C4; cc += 256)
        orow[cc] = (half_t)((vals[cc] - mean) * rstd * g[cc] + b[cc]);
}

// ---------------------------------------------------------------------------
// Windowed attention: one block per (window, head). L=49, head_dim=32.
// QK^T*scale + rel bias + shift mask, fp32 softmax, *V. I/O in f16.
// ---------------------------------------------------------------------------
__global__ __launch_bounds__(256)
void k_attn(const half_t* __restrict__ QKV, half_t* __restrict__ O,
            const float* __restrict__ rel, int C, int heads,
            int Hd, int Wd, int shift)
{
    __shared__ float q[49 * 32], k[49 * 32], v[49 * 32];
    __shared__ float attn[49 * 49];
    __shared__ float lbl[49];

    const int head = blockIdx.x % heads;
    const int win  = blockIdx.x / heads;
    const int nw = Wd / 7;
    const int nhw = (Hd / 7) * nw;
    const int wloc = win % nhw;
    const int wy = wloc / nw, wx = wloc % nw;

    const int tid = threadIdx.x;
    const size_t base = (size_t)win * 49;
    const int C3 = 3 * C;
    const float scale = 0.17677669529663687f;   // 1/sqrt(32)

    for (int idx = tid; idx < 49 * 32; idx += 256) {
        int i = idx >> 5, d = idx & 31;
        size_t roff = (base + i) * (size_t)C3 + head * 32 + d;
        q[idx] = (float)QKV[roff] * scale;
        k[idx] = (float)QKV[roff + C];
        v[idx] = (float)QKV[roff + 2 * C];
    }
    if (tid < 49) {
        if (shift) {
            int py = tid / 7, px = tid % 7;
            int y = wy * 7 + py, x = wx * 7 + px;
            int ay = (y < Hd - 7) ? 0 : ((y < Hd - shift) ? 1 : 2);
            int ax = (x < Wd - 7) ? 0 : ((x < Wd - shift) ? 1 : 2);
            lbl[tid] = (float)(ay * 3 + ax);
        } else {
            lbl[tid] = 0.f;
        }
    }
    __syncthreads();

    for (int idx = tid; idx < 49 * 49; idx += 256) {
        int i = idx / 49, j = idx % 49;
        float s = 0.f;
#pragma unroll
        for (int d = 0; d < 32; ++d) s += q[i * 32 + d] * k[j * 32 + d];
        int iy = i / 7, ix = i % 7, jy = j / 7, jx = j % 7;
        int ridx = (iy - jy + 6) * 13 + (ix - jx + 6);
        s += rel[ridx * heads + head];
        if (lbl[i] != lbl[j]) s += -100.f;
        attn[idx] = s;
    }
    __syncthreads();

    if (tid < 49) {
        float* row = attn + tid * 49;
        float mx = row[0];
        for (int j = 1; j < 49; ++j) mx = fmaxf(mx, row[j]);
        float sm = 0.f;
        for (int j = 0; j < 49; ++j) { float e = __expf(row[j] - mx); row[j] = e; sm += e; }
        float inv = 1.f / sm;
        for (int j = 0; j < 49; ++j) row[j] *= inv;
    }
    __syncthreads();

    for (int idx = tid; idx < 49 * 32; idx += 256) {
        int i = idx >> 5, d = idx & 31;
        float s = 0.f;
        for (int j = 0; j < 49; ++j) s += attn[i * 49 + j] * v[j * 32 + d];
        O[(base + i) * (size_t)C + head * 32 + d] = (half_t)s;
    }
}

// ---------------------------------------------------------------------------
// Final transpose: [img,49,1024] -> [B,T,1024,7,7] flat.
// ---------------------------------------------------------------------------
__global__ __launch_bounds__(256)
void k_out_transpose(const float* __restrict__ Hb, float* __restrict__ Out)
{
    int idx = blockIdx.x * 256 + threadIdx.x;     // 1568*1024
    if (idx >= 1568 * 1024) return;
    int c = idx % 1024;
    int row = idx / 1024;                          // img*49 + pos
    int img = row / 49, pos = row % 49;
    Out[((size_t)img * 1024 + c) * 49 + pos] = Hb[idx];
}

// ---------------------------------------------------------------------------
// Host orchestration
// ---------------------------------------------------------------------------
struct BlockP {
    const float *fc1_b, *fc1_w, *fc2_b, *fc2_w, *ln1_b, *ln1_g, *ln2_b, *ln2_g,
                *proj_b, *proj_w, *qkv_b, *qkv_w, *rel;
};
struct MergeP { const float *ln_b, *ln_g, *red_w; };

static inline dim3 gemm_grid(int M, int N) {
    return dim3((M + TILE_M - 1) / TILE_M, (N + TILE_N - 1) / TILE_N);
}

extern "C" void kernel_launch(void* const* d_in, const int* in_sizes, int n_in,
                              void* d_out, int out_size, void* d_ws, size_t ws_size,
                              hipStream_t stream)
{
    (void)in_sizes; (void)n_in; (void)out_size; (void)ws_size;

    // jax-tree (key-sorted) flattening order of setup_inputs():
    const float* imgs       = (const float*)d_in[0];
    const float* head_b     = (const float*)d_in[1];
    const float* head_w     = (const float*)d_in[2];
    const float* norm_b     = (const float*)d_in[3];
    const float* norm_g     = (const float*)d_in[4];
    const float* patch_b    = (const float*)d_in[5];
    const float* patch_ln_b = (const float*)d_in[6];
    const float* patch_ln_g = (const float*)d_in[7];
    const float* patch_w    = (const float*)d_in[8];

    const int depths[4] = {2, 2, 6, 2};
    BlockP blocks[12];
    MergeP merges[3];
    int p = 9, bi = 0;
    for (int s = 0; s < 4; ++s) {
        for (int j = 0; j < depths[s]; ++j) {
            BlockP& B = blocks[bi++];
            B.fc1_b  = (const float*)d_in[p++];
            B.fc1_w  = (const float*)d_in[p++];
            B.fc2_b  = (const float*)d_in[p++];
            B.fc2_w  = (const float*)d_in[p++];
            B.ln1_b  = (const float*)d_in[p++];
            B.ln1_g  = (const float*)d_in[p++];
            B.ln2_b  = (const float*)d_in[p++];
            B.ln2_g  = (const float*)d_in[p++];
            B.proj_b = (const float*)d_in[p++];
            B.proj_w = (const float*)d_in[p++];
            B.qkv_b  = (const float*)d_in[p++];
            B.qkv_w  = (const float*)d_in[p++];
            B.rel    = (const float*)d_in[p++];
        }
        if (s < 3) {
            merges[s].ln_b  = (const float*)d_in[p++];
            merges[s].ln_g  = (const float*)d_in[p++];
            merges[s].red_w = (const float*)d_in[p++];
        }
    }

    // Workspace carve-up (peak stage-1 sizes; H16 aliases QKV+O region).
    char* ws = (char*)d_ws;
    float*  X    = (float*)ws;                       ws += (size_t)100352 * 96 * 4;  // residual f32
    half_t* A16  = (half_t*)ws;                      ws += (size_t)100352 * 96 * 2;  // LN'd tokens f16
    char*   shr  = ws;                               ws += (size_t)100352 * 384 * 2; // 77,070,336 B
    half_t* QKV  = (half_t*)shr;                                                    // 57,802,752 B
    half_t* O16  = (half_t*)(shr + (size_t)100352 * 288 * 2);                       // 19,267,584 B
    half_t* H16  = (half_t*)shr;                                                    // aliases QKV+O16
    float*  TMP  = (float*)ws;                                                      // head output

    // Patch embed + first LN
    k_patch_embed<<<100352, 96, 0, stream>>>(imgs, patch_w, patch_b,
                                             patch_ln_g, patch_ln_b, X);

    int Hc = 56, Cc = 96;
    bi = 0;
    for (int s = 0; s < 4; ++s) {
        const int heads = 3 << s;
        const int T = 32 * Hc * Hc;
        const int nwin = T / 49;
        for (int j = 0; j < depths[s]; ++j) {
            BlockP& B = blocks[bi++];
            const int shift = (j % 2 == 1 && Hc > 7) ? 3 : 0;

            // LN1 + shift + window partition -> A16 (window-ordered f16)
            k_ln_window<<<T, 256, 0, stream>>>(X, B.ln1_g, B.ln1_b, A16,
                                               Hc, Hc, Cc, shift);
            // qkv = A16 @ qkv_w + b  -> f16
            k_gemm_f16<EPI_F16><<<gemm_grid(T, 3 * Cc), 256, 0, stream>>>(
                A16, B.qkv_w, B.qkv_b, QKV, T, 3 * Cc, Cc, 0, 0, 0);
            // attention per (window, head) -> O16 f16
            k_attn<<<nwin * heads, 256, 0, stream>>>(QKV, O16, B.rel,
                                                     Cc, heads, Hc, Hc, shift);
            // proj with fused un-window + un-shift + residual add into X
            k_gemm_f16<EPI_F32_ADD_WIN><<<gemm_grid(T, Cc), 256, 0, stream>>>(
                O16, B.proj_w, B.proj_b, X, T, Cc, Cc, Hc, Hc, shift);
            // LN2 -> A16 (natural token order, matches X layout)
            k_ln_plain<<<T, 256, 0, stream>>>(X, B.ln2_g, B.ln2_b, A16, Cc);
            // fc1 + GELU -> H16 f16
            k_gemm_f16<EPI_F16_GELU><<<gemm_grid(T, 4 * Cc), 256, 0, stream>>>(
                A16, B.fc1_w, B.fc1_b, H16, T, 4 * Cc, Cc, 0, 0, 0);
            // fc2 accumulated directly into residual X
            k_gemm_f16<EPI_F32_ACC><<<gemm_grid(T, Cc), 256, 0, stream>>>(
                H16, B.fc2_w, B.fc2_b, X, T, Cc, 4 * Cc, 0, 0, 0);
        }
        if (s < 3) {
            const int T2 = T / 4;
            k_merge_ln<<<T2, 256, 0, stream>>>(X, merges[s].ln_g, merges[s].ln_b,
                                               A16, Hc, Hc, Cc);
            k_gemm_f16<EPI_F32><<<gemm_grid(T2, 2 * Cc), 256, 0, stream>>>(
                A16, merges[s].red_w, nullptr, X, T2, 2 * Cc, 4 * Cc, 0, 0, 0);
            Hc >>= 1; Cc <<= 1;
        }
    }

    // Final norm + head + output transpose  (Hc=7, Cc=768, T=1568)
    k_ln_plain<<<1568, 256, 0, stream>>>(X, norm_g, norm_b, A16, 768);
    k_gemm_f16<EPI_F32><<<gemm_grid(1568, 1024), 256, 0, stream>>>(
        A16, head_w, head_b, TMP, 1568, 1024, 768, 0, 0, 0);
    k_out_transpose<<<(1568 * 1024 + 255) / 256, 256, 0, stream>>>(TMP, (float*)d_out);
}
